// WanSelfAttention_34626026340484
// MI455X (gfx1250) — matrix operand order repaired and checked by reading, not compiled, not run on hardware
//
#include <hip/hip_runtime.h>

#define DIM 1536
#define S_TOT 2880
#define HEADS 12
#define HD 128
#define CC 64
#define C1 22
#define C2 21

typedef __attribute__((ext_vector_type(16))) __bf16 v16bf;
typedef __attribute__((ext_vector_type(8)))  float  v8f;
typedef __attribute__((ext_vector_type(4))) unsigned int u32x4;
typedef __attribute__((ext_vector_type(8))) int i32x8;
typedef __attribute__((ext_vector_type(4))) int i32x4;

union Frag16 { v16bf v; unsigned int u[8]; };
union Pack8  { unsigned short s[8]; uint4 q; };

__device__ __forceinline__ unsigned short f2bf(float f) {
  unsigned int u = __float_as_uint(f);
  u += 0x7FFFu + ((u >> 16) & 1u);      // round-to-nearest-even
  return (unsigned short)(u >> 16);
}

__device__ __forceinline__ v8f v8f_zero() {
  v8f z = {0.f, 0.f, 0.f, 0.f, 0.f, 0.f, 0.f, 0.f};
  return z;
}

// A-fragment K index, contiguous order k = 16*hi + 2*v (lane half 'hi', vgpr v, even element)
__device__ __forceinline__ int a_kidx(int hi, int v) {
  return hi * 16 + 2 * v;
}

// ---------------------------------------------------------------------------
// Tensor Data Mover: 2-D tile (bf16 elements) global -> LDS.
// D# per CDNA5 ISA (group0: count/lds/global/type,
// group1: data_size + tensor dims + tile dims + stride). Uniform args only;
// issue from one wave, track with TENSORcnt.
// ---------------------------------------------------------------------------
__device__ __forceinline__ void tdm_load_2d_bf16(unsigned int lds_off,
                                                 const void* gptr,
                                                 unsigned int tile_d0, unsigned int tile_d1,
                                                 unsigned int tdim0, unsigned int tdim1,
                                                 unsigned int stride0) {
  unsigned long long ga = (unsigned long long)gptr;
  u32x4 g0;
  g0[0] = 1u;                                           // count=1 (valid user D#)
  g0[1] = lds_off;                                      // lds_addr (bytes)
  g0[2] = (unsigned int)(ga & 0xFFFFFFFFu);             // global_addr[31:0]
  g0[3] = (unsigned int)((ga >> 32) & 0x01FFFFFFu)      // global_addr[56:32]
          | (2u << 30);                                 // type=2 ("image")
  i32x8 g1;
  g1[0] = (int)(1u << 16);                              // data_size=1 -> 2 bytes
  g1[1] = (int)((tdim0 & 0xFFFFu) << 16);               // tensor_dim0[15:0]
  g1[2] = (int)(((tdim0 >> 16) & 0xFFFFu) | ((tdim1 & 0xFFFFu) << 16));
  g1[3] = (int)(((tdim1 >> 16) & 0xFFFFu) | ((tile_d0 & 0xFFFFu) << 16));
  g1[4] = (int)(tile_d1 & 0xFFFFu);                     // tile_dim1, tile_dim2=0
  g1[5] = (int)stride0;                                 // tensor_dim0_stride[31:0]
  g1[6] = 0;
  g1[7] = 0;
  i32x4 z4 = {0, 0, 0, 0};
#if defined(__clang_major__) && (__clang_major__ >= 23)
  i32x8 z8 = {0, 0, 0, 0, 0, 0, 0, 0};
  __builtin_amdgcn_tensor_load_to_lds(g0, g1, z4, z4, z8, 0);
#else
  __builtin_amdgcn_tensor_load_to_lds(g0, g1, z4, z4, 0);
#endif
}

__device__ __forceinline__ unsigned int lds_offset_of(const void* p) {
  return (unsigned int)(size_t)p;   // generic ptr low 32 bits = LDS offset
}

// ---------------------------------------------------------------------------
// Conversions + axial RoPE frequency grids
// ---------------------------------------------------------------------------
// Transpose + convert one 64x64 tile of W (f32 row-major [k][n]) into
// wt (bf16 [n][k]) so GEMM B-tiles become straight contiguous copies.
__global__ __launch_bounds__(256)
void cvt_wt_kernel(const float* __restrict__ wq, const float* __restrict__ wk,
                   const float* __restrict__ wv, const float* __restrict__ wo,
                   unsigned short* __restrict__ wt) {
  __shared__ __align__(16) unsigned short lds[64 * 72];   // [n][k], pad 8
  const int mat = blockIdx.z;
  const float* W = (mat == 0) ? wq : (mat == 1) ? wk : (mat == 2) ? wv : wo;
  unsigned short* dst = wt + (size_t)mat * DIM * DIM;
  const int k0 = blockIdx.x * 64;
  const int n0 = blockIdx.y * 64;
  const int tid = threadIdx.x;
#pragma unroll
  for (int it = 0; it < 16; ++it) {
    int idx = it * 256 + tid;
    int r = idx >> 6, c = idx & 63;                       // r=k, c=n
    lds[c * 72 + r] = f2bf(W[(size_t)(k0 + r) * DIM + n0 + c]);
  }
  __syncthreads();
#pragma unroll
  for (int it = 0; it < 2; ++it) {
    int e = (it * 256 + tid) * 8;
    int rr = e >> 6, cc = e & 63;                         // rr=n, cc=k
    Pack8 pk;
#pragma unroll
    for (int j = 0; j < 8; ++j) pk.s[j] = lds[rr * 72 + cc + j];
    *(uint4*)(dst + (size_t)(n0 + rr) * DIM + k0 + cc) = pk.q;
  }
}

__global__ void cvt_x_kernel(const float* __restrict__ x, unsigned short* __restrict__ xbf) {
  const size_t N = (size_t)S_TOT * DIM;
  const size_t stride = (size_t)gridDim.x * blockDim.x;
  for (size_t i = (size_t)blockIdx.x * blockDim.x + threadIdx.x; i < N; i += stride)
    xbf[i] = f2bf(x[i]);
}

__global__ void freqs_kernel(const int* __restrict__ grid_sizes,
                             const float* __restrict__ fcos, const float* __restrict__ fsin,
                             float* __restrict__ cosg, float* __restrict__ sing) {
  int idx = blockIdx.x * blockDim.x + threadIdx.x;
  if (idx >= S_TOT * CC) return;
  int sIdx = idx / CC, c = idx - sIdx * CC;
  int H = grid_sizes[1], W = grid_sizes[2];
  int f = sIdx / (H * W);
  int rem = sIdx - f * (H * W);
  int hh = rem / W, ww = rem - hh * W;
  int pos = (c < C1) ? f : ((c < C1 + C2) ? hh : ww);
  cosg[idx] = fcos[pos * CC + c];
  sing[idx] = fsin[pos * CC + c];
}

// ---------------------------------------------------------------------------
// QKV projection GEMM: TDM-fed, double-buffered, 64x64 tile, 4 waves.
// z = 0:q(f32) 1:k(f32) 2:v -> written TRANSPOSED as bf16 vt[dim][seq]
// ---------------------------------------------------------------------------
__global__ __launch_bounds__(128)
void gemm_qkv_kernel(const unsigned short* __restrict__ xbf,
                     const unsigned short* __restrict__ wt,
                     const float* __restrict__ bq, const float* __restrict__ bk,
                     const float* __restrict__ bv,
                     float* __restrict__ qraw, float* __restrict__ kraw,
                     unsigned short* __restrict__ vt) {
  __shared__ __align__(16) unsigned short ldsA[2][64 * 32];   // [m][k]
  __shared__ __align__(16) unsigned short ldsBt[2][64 * 32];  // [n][k]
  const int tid = threadIdx.x;
  const int lane = tid & 31;
  const int wid = tid >> 5;
  const int n0blk = blockIdx.x * 64;
  const int m0blk = blockIdx.y * 64;
  const int mat = blockIdx.z;
  const unsigned short* Wmat = wt + (size_t)mat * DIM * DIM;
  const int m = lane & 15, hi = lane >> 4;
  const int NK = DIM / 32;

  v8f acc[4];
#pragma unroll
  for (int t = 0; t < 4; ++t) acc[t] = v8f_zero();

  if (tid < 32) {
    tdm_load_2d_bf16(lds_offset_of(&ldsA[0][0]), xbf + (size_t)m0blk * DIM,
                     32, 64, DIM, S_TOT, DIM);
    tdm_load_2d_bf16(lds_offset_of(&ldsBt[0][0]), Wmat + (size_t)n0blk * DIM,
                     32, 64, DIM, DIM, DIM);
    __builtin_amdgcn_s_wait_tensorcnt(0);
  }
  __syncthreads();

  for (int ks = 0; ks < NK; ++ks) {
    const int st = ks & 1;
    if (ks + 1 < NK && tid < 32) {
      int k1 = (ks + 1) * 32;
      tdm_load_2d_bf16(lds_offset_of(&ldsA[1 - st][0]), xbf + (size_t)m0blk * DIM + k1,
                       32, 64, DIM, S_TOT, DIM);
      tdm_load_2d_bf16(lds_offset_of(&ldsBt[1 - st][0]), Wmat + (size_t)n0blk * DIM + k1,
                       32, 64, DIM, DIM, DIM);
    }
    Frag16 a;
#pragma unroll
    for (int v = 0; v < 8; ++v)
      a.u[v] = *(const unsigned int*)(&ldsA[st][(wid * 16 + m) * 32 + a_kidx(hi, v)]);
#pragma unroll
    for (int t = 0; t < 4; ++t) {
      Frag16 b;
      int n = t * 16 + m;
#pragma unroll
      for (int v = 0; v < 8; ++v)
        b.u[v] = *(const unsigned int*)(&ldsBt[st][n * 32 + hi * 16 + 2 * v]);
      acc[t] = __builtin_amdgcn_wmma_f32_16x16x32_bf16(false, a.v, false, b.v,
                                                       (short)0, acc[t], false, false);
    }
    if (tid < 32) __builtin_amdgcn_s_wait_tensorcnt(0);
    __syncthreads();
  }

  const int grow = m0blk + wid * 16;
  if (mat == 2) {
    // V epilogue: bias + bf16, stored transposed vt[n][row]; 8 consecutive rows
    // per lane are contiguous -> one 16B store per tile.
#pragma unroll
    for (int t = 0; t < 4; ++t) {
      int n = n0blk + t * 16 + m;
      float bval = bv[n];
      Pack8 pk;
#pragma unroll
      for (int v = 0; v < 8; ++v) pk.s[v] = f2bf(acc[t][v] + bval);
      *(uint4*)(vt + (size_t)n * S_TOT + grow + 8 * hi) = pk.q;
    }
  } else {
    float* dst = (mat == 0) ? qraw : kraw;
    const float* bias = (mat == 0) ? bq : bk;
#pragma unroll
    for (int t = 0; t < 4; ++t) {
      int n = n0blk + t * 16 + m;
      float bval = bias[n];
#pragma unroll
      for (int v = 0; v < 8; ++v)
        dst[(size_t)(grow + v + 8 * hi) * DIM + n] = acc[t][v] + bval;
    }
  }
}

// ---------------------------------------------------------------------------
// Output projection GEMM: out = o_bf16 @ Wo + bo (f32), TDM-fed
// ---------------------------------------------------------------------------
__global__ __launch_bounds__(128)
void gemm_out_kernel(const unsigned short* __restrict__ obf,
                     const unsigned short* __restrict__ Wt,
                     const float* __restrict__ bo, float* __restrict__ out) {
  __shared__ __align__(16) unsigned short ldsA[2][64 * 32];
  __shared__ __align__(16) unsigned short ldsBt[2][64 * 32];
  const int tid = threadIdx.x;
  const int lane = tid & 31;
  const int wid = tid >> 5;
  const int n0blk = blockIdx.x * 64;
  const int m0blk = blockIdx.y * 64;
  const int m = lane & 15, hi = lane >> 4;
  const int NK = DIM / 32;

  v8f acc[4];
#pragma unroll
  for (int t = 0; t < 4; ++t) acc[t] = v8f_zero();

  if (tid < 32) {
    tdm_load_2d_bf16(lds_offset_of(&ldsA[0][0]), obf + (size_t)m0blk * DIM,
                     32, 64, DIM, S_TOT, DIM);
    tdm_load_2d_bf16(lds_offset_of(&ldsBt[0][0]), Wt + (size_t)n0blk * DIM,
                     32, 64, DIM, DIM, DIM);
    __builtin_amdgcn_s_wait_tensorcnt(0);
  }
  __syncthreads();

  for (int ks = 0; ks < NK; ++ks) {
    const int st = ks & 1;
    if (ks + 1 < NK && tid < 32) {
      int k1 = (ks + 1) * 32;
      tdm_load_2d_bf16(lds_offset_of(&ldsA[1 - st][0]), obf + (size_t)m0blk * DIM + k1,
                       32, 64, DIM, S_TOT, DIM);
      tdm_load_2d_bf16(lds_offset_of(&ldsBt[1 - st][0]), Wt + (size_t)n0blk * DIM + k1,
                       32, 64, DIM, DIM, DIM);
    }
    Frag16 a;
#pragma unroll
    for (int v = 0; v < 8; ++v)
      a.u[v] = *(const unsigned int*)(&ldsA[st][(wid * 16 + m) * 32 + a_kidx(hi, v)]);
#pragma unroll
    for (int t = 0; t < 4; ++t) {
      Frag16 b;
      int n = t * 16 + m;
#pragma unroll
      for (int v = 0; v < 8; ++v)
        b.u[v] = *(const unsigned int*)(&ldsBt[st][n * 32 + hi * 16 + 2 * v]);
      acc[t] = __builtin_amdgcn_wmma_f32_16x16x32_bf16(false, a.v, false, b.v,
                                                       (short)0, acc[t], false, false);
    }
    if (tid < 32) __builtin_amdgcn_s_wait_tensorcnt(0);
    __syncthreads();
  }

  const int grow = m0blk + wid * 16;
#pragma unroll
  for (int t = 0; t < 4; ++t) {
    int n = n0blk + t * 16 + m;
    float bval = bo[n];
#pragma unroll
    for (int v = 0; v < 8; ++v)
      out[(size_t)(grow + v + 8 * hi) * DIM + n] = acc[t][v] + bval;
  }
}

// ---------------------------------------------------------------------------
// RMSNorm (full 1536-dim) + axial RoPE, f32 -> bf16
// ---------------------------------------------------------------------------
__global__ __launch_bounds__(256)
void rmsnorm_rope_kernel(const float* __restrict__ qraw, const float* __restrict__ kraw,
                         const float* __restrict__ gq, const float* __restrict__ gk,
                         const float* __restrict__ cosg, const float* __restrict__ sing,
                         unsigned short* __restrict__ qbf, unsigned short* __restrict__ kbf) {
  const int s = blockIdx.x;
  const int which = blockIdx.y;
  const float* src = ((which == 0) ? qraw : kraw) + (size_t)s * DIM;
  const float* g = (which == 0) ? gq : gk;
  unsigned short* dst = ((which == 0) ? qbf : kbf) + (size_t)s * DIM;
  const int tid = threadIdx.x;

  float vals[6];
  float ss = 0.f;
#pragma unroll
  for (int i = 0; i < 6; ++i) {
    vals[i] = src[tid * 6 + i];
    ss += vals[i] * vals[i];
  }
#pragma unroll
  for (int off = 16; off > 0; off >>= 1) ss += __shfl_xor(ss, off, 32);
  __shared__ float red[8];
  if ((tid & 31) == 0) red[tid >> 5] = ss;
  __syncthreads();
  float tot = 0.f;
#pragma unroll
  for (int i = 0; i < 8; ++i) tot += red[i];
  const float rinv = rsqrtf(tot / (float)DIM + 1e-6f);

#pragma unroll
  for (int p = 0; p < 3; ++p) {
    int col = tid * 6 + p * 2;
    int i = (col & (HD - 1)) >> 1;            // RoPE pair index within head
    float c = cosg[(size_t)s * CC + i];
    float sn = sing[(size_t)s * CC + i];
    float yr = vals[p * 2] * rinv * g[col];
    float yi = vals[p * 2 + 1] * rinv * g[col + 1];
    dst[col]     = f2bf(yr * c - yi * sn);
    dst[col + 1] = f2bf(yr * sn + yi * c);
  }
}

// ---------------------------------------------------------------------------
// Flash attention: block = (head, 64 q rows), 4 waves x 16 rows, 32-key chunks.
// K chunk + pre-transposed V chunk staged by TDM, double-buffered.
// ---------------------------------------------------------------------------
__global__ __launch_bounds__(128)
void attn_kernel(const unsigned short* __restrict__ qbf,
                 const unsigned short* __restrict__ kbf,
                 const unsigned short* __restrict__ vt,
                 unsigned short* __restrict__ obf) {
  __shared__ __align__(16) unsigned short ldsK[2][32 * HD];     // [key][dim]
  __shared__ __align__(16) unsigned short ldsVt[2][HD * 32];    // [dim][key]
  __shared__ __align__(16) unsigned short ldsP[4][16 * 32];     // per-wave P
  const int tid = threadIdx.x;
  const int lane = tid & 31;
  const int wid = tid >> 5;
  const int h = blockIdx.x;
  const int q0 = blockIdx.y * 64;
  const int m = lane & 15, hi = lane >> 4;
  const int qrow = q0 + wid * 16 + m;
  const size_t hoff = (size_t)h * HD;
  const float scale = 0.08838834764831845f;  // 1/sqrt(128)
  const int NCH = S_TOT / 32;

  // Q fragments (resident): 4 K-steps of 32 over HD=128
  Frag16 qa[4];
#pragma unroll
  for (int s = 0; s < 4; ++s)
#pragma unroll
    for (int v = 0; v < 8; ++v)
      qa[s].u[v] = *(const unsigned int*)(qbf + (size_t)qrow * DIM + hoff + s * 32 + a_kidx(hi, v));

  v8f o[8];
#pragma unroll
  for (int t = 0; t < 8; ++t) o[t] = v8f_zero();
  float rmax[8], rsum[8];
#pragma unroll
  for (int v = 0; v < 8; ++v) { rmax[v] = -1e30f; rsum[v] = 0.f; }

  if (tid < 32) {
    tdm_load_2d_bf16(lds_offset_of(&ldsK[0][0]), kbf + hoff,
                     HD, 32, DIM, S_TOT, DIM);
    tdm_load_2d_bf16(lds_offset_of(&ldsVt[0][0]), vt + hoff * S_TOT,
                     32, HD, S_TOT, DIM, S_TOT);
    __builtin_amdgcn_s_wait_tensorcnt(0);
  }
  __syncthreads();

  for (int i = 0; i < NCH; ++i) {
    const int st = i & 1;
    if (i + 1 < NCH && tid < 32) {
      int kc1 = (i + 1) * 32;
      tdm_load_2d_bf16(lds_offset_of(&ldsK[1 - st][0]), kbf + (size_t)kc1 * DIM + hoff,
                       HD, 32, DIM, S_TOT, DIM);
      tdm_load_2d_bf16(lds_offset_of(&ldsVt[1 - st][0]), vt + hoff * S_TOT + kc1,
                       32, HD, S_TOT, DIM, S_TOT);
    }

    // scores: 16 q rows x 32 keys  (K^T fragments are contiguous u32 LDS reads)
    v8f sc[2];
    sc[0] = v8f_zero(); sc[1] = v8f_zero();
#pragma unroll
    for (int s = 0; s < 4; ++s)
#pragma unroll
      for (int t = 0; t < 2; ++t) {
        Frag16 b;
        int n = t * 16 + m;
#pragma unroll
        for (int v = 0; v < 8; ++v)
          b.u[v] = *(const unsigned int*)(&ldsK[st][n * HD + s * 32 + hi * 16 + 2 * v]);
        sc[t] = __builtin_amdgcn_wmma_f32_16x16x32_bf16(false, qa[s].v, false, b.v,
                                                        (short)0, sc[t], false, false);
      }

    // online softmax (stats per C-matrix row: row = v + 8*hi, replicated in half)
    float corr[8];
#pragma unroll
    for (int v = 0; v < 8; ++v) {
      float a0 = sc[0][v] * scale, a1 = sc[1][v] * scale;
      float cm = fmaxf(a0, a1);
#pragma unroll
      for (int off = 1; off < 16; off <<= 1) cm = fmaxf(cm, __shfl_xor(cm, off, 32));
      float nm = fmaxf(rmax[v], cm);
      float c = __expf(rmax[v] - nm);
      float p0 = __expf(a0 - nm);
      float p1 = __expf(a1 - nm);
      float ps = p0 + p1;
#pragma unroll
      for (int off = 1; off < 16; off <<= 1) ps += __shfl_xor(ps, off, 32);
      rsum[v] = rsum[v] * c + ps;
      rmax[v] = nm;
      corr[v] = c;
      ldsP[wid][(v + 8 * hi) * 32 + m]      = f2bf(p0);
      ldsP[wid][(v + 8 * hi) * 32 + 16 + m] = f2bf(p1);
    }
#pragma unroll
    for (int t = 0; t < 8; ++t)
#pragma unroll
      for (int v = 0; v < 8; ++v) o[t][v] *= corr[v];

    // O += P @ V  (V^T fragments are contiguous u32 LDS reads)
    Frag16 pa;
#pragma unroll
    for (int v = 0; v < 8; ++v)
      pa.u[v] = *(const unsigned int*)(&ldsP[wid][m * 32 + a_kidx(hi, v)]);
#pragma unroll
    for (int t = 0; t < 8; ++t) {
      Frag16 b;
#pragma unroll
      for (int v = 0; v < 8; ++v)
        b.u[v] = *(const unsigned int*)(&ldsVt[st][(t * 16 + m) * 32 + hi * 16 + 2 * v]);
      o[t] = __builtin_amdgcn_wmma_f32_16x16x32_bf16(false, pa.v, false, b.v,
                                                     (short)0, o[t], false, false);
    }

    if (tid < 32) __builtin_amdgcn_s_wait_tensorcnt(0);
    __syncthreads();
  }

#pragma unroll
  for (int t = 0; t < 8; ++t) {
    int n = t * 16 + m;
#pragma unroll
    for (int v = 0; v < 8; ++v) {
      int row = q0 + wid * 16 + v + 8 * hi;
      obf[(size_t)row * DIM + hoff + n] = f2bf(o[t][v] / rsum[v]);
    }
  }
}

// ---------------------------------------------------------------------------
extern "C" void kernel_launch(void* const* d_in, const int* in_sizes, int n_in,
                              void* d_out, int out_size, void* d_ws, size_t ws_size,
                              hipStream_t stream) {
  const float* x          = (const float*)d_in[0];
  const int*   grid_sizes = (const int*)d_in[2];
  const float* fcos       = (const float*)d_in[3];
  const float* fsin       = (const float*)d_in[4];
  const float* Wq = (const float*)d_in[5];
  const float* bq = (const float*)d_in[6];
  const float* Wk = (const float*)d_in[7];
  const float* bk = (const float*)d_in[8];
  const float* Wv = (const float*)d_in[9];
  const float* bv = (const float*)d_in[10];
  const float* Wo = (const float*)d_in[11];
  const float* bo = (const float*)d_in[12];
  const float* gq = (const float*)d_in[13];
  const float* gk = (const float*)d_in[14];
  float* out = (float*)d_out;

  char* ws = (char*)d_ws;
  size_t off = 0;
  auto alloc = [&](size_t bytes) -> void* {
    void* p = ws + off;
    off += (bytes + 255) & ~(size_t)255;
    return p;
  };
  const size_t WN = (size_t)DIM * DIM;
  const size_t XN = (size_t)S_TOT * DIM;
  unsigned short* wt   = (unsigned short*)alloc(4 * WN * 2);  // [mat][n][k] bf16
  unsigned short* xbf  = (unsigned short*)alloc(XN * 2);
  float*          qraw = (float*)alloc(XN * 4);
  float*          kraw = (float*)alloc(XN * 4);
  unsigned short* vt   = (unsigned short*)alloc(XN * 2);      // [dim][seq] bf16
  unsigned short* qbf  = (unsigned short*)alloc(XN * 2);
  unsigned short* kbf  = (unsigned short*)alloc(XN * 2);
  unsigned short* obf  = (unsigned short*)alloc(XN * 2);
  float*          cosg = (float*)alloc((size_t)S_TOT * CC * 4);
  float*          sing = (float*)alloc((size_t)S_TOT * CC * 4);

  cvt_wt_kernel<<<dim3(DIM / 64, DIM / 64, 4), 256, 0, stream>>>(Wq, Wk, Wv, Wo, wt);
  cvt_x_kernel<<<2048, 256, 0, stream>>>(x, xbf);
  freqs_kernel<<<(S_TOT * CC + 255) / 256, 256, 0, stream>>>(grid_sizes, fcos, fsin, cosg, sing);
  gemm_qkv_kernel<<<dim3(DIM / 64, S_TOT / 64, 3), 128, 0, stream>>>(
      xbf, wt, bq, bk, bv, qraw, kraw, vt);
  rmsnorm_rope_kernel<<<dim3(S_TOT, 2), 256, 0, stream>>>(
      qraw, kraw, gq, gk, cosg, sing, qbf, kbf);
  attn_kernel<<<dim3(HEADS, S_TOT / 64), 128, 0, stream>>>(qbf, kbf, vt, obf);
  gemm_out_kernel<<<dim3(DIM / 64, S_TOT / 64), 128, 0, stream>>>(
      obf, wt + 3 * WN, bo, out);
}
